// GCN_13898514169996
// MI455X (gfx1250) — matrix operand order repaired and checked by reading, not compiled, not run on hardware
//
#include <hip/hip_runtime.h>

typedef __attribute__((ext_vector_type(16))) __bf16 v16bf;
typedef __attribute__((ext_vector_type(8)))  float  v8f;

#define D_IN  512
#define D_HID 16
#define D_OUT 7

// ---------------------------------------------------------------------------
// Load one A-fragment row slice for v_wmma_f32_16x16x32_bf16.
// Contiguous K order (matches B): this lane needs K = kbase..kbase+15
// (kbase already includes the half-wave +16 shift).
// ---------------------------------------------------------------------------
__device__ __forceinline__ v16bf load_a_frag(const float* __restrict__ xrow, int kbase) {
  const float4* p0 = (const float4*)(xrow + kbase);
  const float4* p1 = (const float4*)(xrow + kbase + 8);
  float4 f0 = p0[0], f1 = p0[1];
  float4 f2 = p1[0], f3 = p1[1];
  v16bf a;
  a[0]  = (__bf16)f0.x; a[1]  = (__bf16)f0.y; a[2]  = (__bf16)f0.z; a[3]  = (__bf16)f0.w;
  a[4]  = (__bf16)f1.x; a[5]  = (__bf16)f1.y; a[6]  = (__bf16)f1.z; a[7]  = (__bf16)f1.w;
  a[8]  = (__bf16)f2.x; a[9]  = (__bf16)f2.y; a[10] = (__bf16)f2.z; a[11] = (__bf16)f2.w;
  a[12] = (__bf16)f3.x; a[13] = (__bf16)f3.y; a[14] = (__bf16)f3.z; a[15] = (__bf16)f3.w;
  return a;
}

// W1 [512 x 16] f32  ->  Wt [16 x 512] bf16 (column n of W1 becomes row n of Wt)
__global__ void gcn_prep_w(const float* __restrict__ W1, __bf16* __restrict__ Wt) {
  int t = blockIdx.x * blockDim.x + threadIdx.x;
  if (t >= D_HID * D_IN) return;
  int n = t >> 9;          // output row (= W1 column)
  int k = t & (D_IN - 1);  // output col (= W1 row)
  Wt[t] = (__bf16)W1[k * D_HID + n];
}

__global__ void gcn_deg_init(float* __restrict__ deg, int n) {
  int i = blockIdx.x * blockDim.x + threadIdx.x;
  if (i < n) deg[i] = 1.0f;  // self-loop
}

__global__ void gcn_deg_edges(const int* __restrict__ dst, float* __restrict__ deg, int e) {
  int i = blockIdx.x * blockDim.x + threadIdx.x;
  if (i < e) atomicAdd(&deg[dst[i]], 1.0f);
}

__global__ void gcn_dinv(float* __restrict__ deg, int n) {
  int i = blockIdx.x * blockDim.x + threadIdx.x;
  if (i < n) deg[i] = rsqrtf(deg[i]);  // deg >= 1 always (self-loop)
}

// ---------------------------------------------------------------------------
// a1 = dinv * (X @ W1); agg1 initialized to a1 (the self-loop contribution).
// One wave computes two 16x16 output tiles (rows row0..row0+31), sharing B.
// ---------------------------------------------------------------------------
__global__ __launch_bounds__(256, 1) void gcn_gemm1_wmma(
    const float* __restrict__ X, const __bf16* __restrict__ Wt,
    const float* __restrict__ dinv, float* __restrict__ a1,
    float* __restrict__ agg1, int n) {
  const int lane = threadIdx.x & 31;
  const int wave = blockIdx.x * (blockDim.x >> 5) + (threadIdx.x >> 5);
  const int row0 = wave * 32;
  if (row0 >= n) return;

  const int m    = lane & 15;
  const int half = lane >> 4;

  const float*  x0   = X  + (size_t)(row0 + m) * D_IN;
  const float*  x1   = X  + (size_t)(row0 + 16 + m) * D_IN;
  const __bf16* wrow = Wt + (size_t)m * D_IN + half * 16;  // B: lane holds col m, K half

  v8f c0 = {};
  v8f c1 = {};
#pragma unroll
  for (int k0 = 0; k0 < D_IN; k0 += 32) {
    v16bf b   = *(const v16bf*)(wrow + k0);          // 32B contiguous, aligned
    v16bf aa0 = load_a_frag(x0, k0 + half * 16);
    v16bf aa1 = load_a_frag(x1, k0 + half * 16);
    c0 = __builtin_amdgcn_wmma_f32_16x16x32_bf16(false, aa0, false, b, (short)0, c0, false, false);
    c1 = __builtin_amdgcn_wmma_f32_16x16x32_bf16(false, aa1, false, b, (short)0, c1, false, false);
  }

#pragma unroll
  for (int r = 0; r < 8; ++r) {
    int M = half * 8 + r;  // C/D layout: lane<16 -> M=r, lane>=16 -> M=8+r; N = lane&15
    {
      int row = row0 + M;
      float v = c0[r] * dinv[row];
      size_t idx = (size_t)row * D_HID + m;
      a1[idx] = v; agg1[idx] = v;
    }
    {
      int row = row0 + 16 + M;
      float v = c1[r] * dinv[row];
      size_t idx = (size_t)row * D_HID + m;
      a1[idx] = v; agg1[idx] = v;
    }
  }
}

// Layer-1 scatter-add: lane = (edge, feature dim). a1/agg1 live in L2 (6.4MB).
__global__ void gcn_edge_agg16(const int* __restrict__ src, const int* __restrict__ dst,
                               const float* __restrict__ a1, float* __restrict__ agg1, int e) {
  int t = blockIdx.x * blockDim.x + threadIdx.x;
  int edge = t >> 4;
  if (edge >= e) return;
  int d = t & 15;
  int s = src[edge], q = dst[edge];
  atomicAdd(&agg1[(size_t)q * D_HID + d], a1[(size_t)s * D_HID + d]);
}

// z1 = relu(dinv*agg1 + b1); h2 = z1 @ W2; a2 = dinv*h2; agg2 init = a2.
__global__ void gcn_layer2_node(const float* __restrict__ agg1, const float* __restrict__ dinv,
                                const float* __restrict__ b1, const float* __restrict__ W2,
                                float* __restrict__ a2, float* __restrict__ agg2, int n) {
  int i = blockIdx.x * blockDim.x + threadIdx.x;
  if (i >= n) return;
  float di = dinv[i];
  float z[D_HID];
#pragma unroll
  for (int d = 0; d < D_HID; ++d) {
    float v = di * agg1[(size_t)i * D_HID + d] + b1[d];
    z[d] = v > 0.0f ? v : 0.0f;
  }
#pragma unroll
  for (int j = 0; j < D_OUT; ++j) {
    float acc = 0.0f;
#pragma unroll
    for (int d = 0; d < D_HID; ++d) acc = fmaf(z[d], W2[d * D_OUT + j], acc);
    float v = di * acc;
    a2[(size_t)i * 8 + j]   = v;
    agg2[(size_t)i * 8 + j] = v;
  }
}

// Layer-2 scatter-add: lane = (edge, dim<8); dim 7 idle.
__global__ void gcn_edge_agg8(const int* __restrict__ src, const int* __restrict__ dst,
                              const float* __restrict__ a2, float* __restrict__ agg2, int e) {
  int t = blockIdx.x * blockDim.x + threadIdx.x;
  int edge = t >> 3;
  if (edge >= e) return;
  int d = t & 7;
  if (d >= D_OUT) return;
  int s = src[edge], q = dst[edge];
  atomicAdd(&agg2[(size_t)q * 8 + d], a2[(size_t)s * 8 + d]);
}

// out = log_softmax(dinv*agg2 + b2)
__global__ void gcn_finalize(const float* __restrict__ agg2, const float* __restrict__ dinv,
                             const float* __restrict__ b2, float* __restrict__ out, int n) {
  int i = blockIdx.x * blockDim.x + threadIdx.x;
  if (i >= n) return;
  float di = dinv[i];
  float v[D_OUT];
  float mx = -3.402823466e+38f;
#pragma unroll
  for (int j = 0; j < D_OUT; ++j) {
    v[j] = di * agg2[(size_t)i * 8 + j] + b2[j];
    mx = fmaxf(mx, v[j]);
  }
  float s = 0.0f;
#pragma unroll
  for (int j = 0; j < D_OUT; ++j) s += expf(v[j] - mx);
  float l = logf(s);
#pragma unroll
  for (int j = 0; j < D_OUT; ++j) out[(size_t)i * D_OUT + j] = v[j] - mx - l;
}

extern "C" void kernel_launch(void* const* d_in, const int* in_sizes, int n_in,
                              void* d_out, int out_size, void* d_ws, size_t ws_size,
                              hipStream_t stream) {
  const float* X  = (const float*)d_in[0];
  const int*   ei = (const int*)d_in[1];
  const float* W1 = (const float*)d_in[2];
  const float* b1 = (const float*)d_in[3];
  const float* W2 = (const float*)d_in[4];
  const float* b2 = (const float*)d_in[5];

  const int n = in_sizes[0] / D_IN;  // 100000
  const int e = in_sizes[1] / 2;     // 3200000
  const int* src = ei;
  const int* dst = ei + e;
  float* out = (float*)d_out;

  // Carve workspace (256B-aligned chunks).
  char* ws = (char*)d_ws;
  auto carve = [&](size_t bytes) {
    char* p = ws;
    ws += (bytes + 255) & ~(size_t)255;
    return p;
  };
  __bf16* Wt   = (__bf16*)carve((size_t)D_HID * D_IN * sizeof(__bf16)); // 16 KB
  float*  deg  = (float*)carve((size_t)n * sizeof(float));              // -> becomes dinv
  float*  a1   = (float*)carve((size_t)n * D_HID * sizeof(float));      // 6.4 MB
  float*  agg1 = (float*)carve((size_t)n * D_HID * sizeof(float));      // 6.4 MB
  float*  a2   = (float*)carve((size_t)n * 8 * sizeof(float));          // 3.2 MB
  float*  agg2 = (float*)carve((size_t)n * 8 * sizeof(float));          // 3.2 MB

  const int B = 256;

  gcn_prep_w<<<(D_HID * D_IN + B - 1) / B, B, 0, stream>>>(W1, Wt);
  gcn_deg_init<<<(n + B - 1) / B, B, 0, stream>>>(deg, n);
  gcn_deg_edges<<<(e + B - 1) / B, B, 0, stream>>>(dst, deg, e);
  gcn_dinv<<<(n + B - 1) / B, B, 0, stream>>>(deg, n);

  int tiles32 = (n + 31) / 32;                 // 32 rows per wave
  int wavesPerBlock = B / 32;
  gcn_gemm1_wmma<<<(tiles32 + wavesPerBlock - 1) / wavesPerBlock, B, 0, stream>>>(
      X, Wt, deg, a1, agg1, n);

  gcn_edge_agg16<<<((long long)e * 16 + B - 1) / B, B, 0, stream>>>(src, dst, a1, agg1, e);
  gcn_layer2_node<<<(n + B - 1) / B, B, 0, stream>>>(agg1, deg, b1, W2, a2, agg2, n);
  gcn_edge_agg8<<<((long long)e * 8 + B - 1) / B, B, 0, stream>>>(src, dst, a2, agg2, e);
  gcn_finalize<<<(n + B - 1) / B, B, 0, stream>>>(agg2, deg, b2, out, n);
}
